// TiSSD_7713761264334
// MI455X (gfx1250) — matrix operand order repaired and checked by reading, not compiled, not run on hardware
//
#include <hip/hip_runtime.h>
#include <hip/hip_bf16.h>

typedef __attribute__((ext_vector_type(16))) _Float16 v16h;
typedef __attribute__((ext_vector_type(8)))  _Float16 v8h;
typedef __attribute__((ext_vector_type(8)))  float    v8f;
typedef __attribute__((ext_vector_type(4)))  unsigned int u32x4;
typedef __attribute__((ext_vector_type(8)))  int      i32x8;
typedef __attribute__((ext_vector_type(4)))  int      i32x4;

#define B_      2
#define SEQ_    4096
#define NH_     24
#define HD_     64
#define DSSM_   1536
#define CONVD_  1792
#define DPROJ_  3352
#define DMODEL_ 768
#define NC_     16
#define CH_     256

__device__ __forceinline__ float siluf(float x) { return x / (1.f + __expf(-x)); }

__device__ __forceinline__ v8f wmma16(v16h a, v16h b, v8f c) {
  return __builtin_amdgcn_wmma_f32_16x16x32_f16(false, a, false, b, (short)0, c, false, false);
}

// A-matrix fragment (16x32 f16, M x K): lane L holds row (L&15),
// K = 16*(L>>4) + {0..15} (contiguous, matching load_b).
__device__ __forceinline__ v16h load_a(const _Float16* base, int ld, int row, int rmax, int k, int lane) {
  int r = row + (lane & 15); if (r > rmax) r = rmax;
  const _Float16* p = base + (size_t)r * ld + k + ((lane >> 4) << 4);
  v8h lo = *(const v8h*)p;
  v8h hi = *(const v8h*)(p + 8);
  v16h f;
#pragma unroll
  for (int i = 0; i < 8; ++i) { f[i] = lo[i]; f[i + 8] = hi[i]; }
  return f;
}

// B-matrix fragment (32x16 f16, K x N) from row-major "Bt" (N x K):
// lane L holds column N=(L&15), K = 16*(L>>4) + {0..15} -- one contiguous 32B run.
__device__ __forceinline__ v16h load_b(const _Float16* base, int ld, int row, int rmax, int k, int lane) {
  int r = row + (lane & 15); if (r > rmax) r = rmax;
  const _Float16* p = base + (size_t)r * ld + k + ((lane >> 4) << 4);
  return *(const v16h*)p;
}

// Tensor Data Mover: 2-D tile (rows x cols_h f16, row stride stride_h) -> LDS.
// D# per CDNA5 ISA ch.8: group0 = {count|flags, lds_addr, global_addr, type=2},
// group1 = {mask/data_size, tensor_dim0/1, tile_dim0/1/2, dim0_stride}.
__device__ __forceinline__ void tdm_load_2d(unsigned lds_off, const void* gaddr,
                                            int rows, int cols_h, int stride_h) {
  unsigned long long ga = (unsigned long long)gaddr;
  u32x4 g0;
  g0[0] = 1u;                                           // count=1, user descriptor
  g0[1] = lds_off;                                      // lds_addr [63:32]
  g0[2] = (unsigned)(ga & 0xFFFFFFFFull);               // global_addr low
  g0[3] = (unsigned)((ga >> 32) & 0x01FFFFFFull)        // global_addr[56:32]
        | 0x80000000u;                                  // type=2 ("image") [127:126]
  i32x8 g1;
  g1[0] = 1 << 16;                                      // workgroup_mask=0, data_size=2B
  g1[1] = cols_h << 16;                                 // tensor_dim0[15:0] @ bits 63:48
  g1[2] = rows << 16;                                   // tensor_dim1[15:0] @ bits 95:80
  g1[3] = cols_h << 16;                                 // tile_dim0 @ bits 127:112
  g1[4] = rows & 0xFFFF;                                // tile_dim1 @ 143:128, tile_dim2=0
  g1[5] = stride_h;                                     // tensor_dim0_stride low32
  g1[6] = 0;
  g1[7] = 0;
  i32x4 z4 = {0, 0, 0, 0};
#if __clang_major__ >= 23
  i32x8 z8 = {0, 0, 0, 0, 0, 0, 0, 0};
  __builtin_amdgcn_tensor_load_to_lds(g0, g1, z4, z4, z8, 0);
#else
  __builtin_amdgcn_tensor_load_to_lds(g0, g1, z4, z4, 0);
#endif
}

__global__ void k_cvt(const float* __restrict__ in, _Float16* __restrict__ out, long n) {
  long i = (long)blockIdx.x * blockDim.x + threadIdx.x;
  long st = (long)gridDim.x * blockDim.x;
  for (; i < n; i += st) out[i] = (_Float16)in[i];
}

// C[M,N] = A[M,K] * Bt[N,K]^T + bias, optional silu. Block = 8 waves, 128x64 tile,
// each wave owns a 32x32 tile (2x2 WMMA accumulators), K-step 32.
__global__ __launch_bounds__(256) void k_gemm(const _Float16* __restrict__ A,
    const _Float16* __restrict__ Bt, const float* __restrict__ bias,
    float* __restrict__ C, int M, int N, int K, int act) {
  int lane = threadIdx.x & 31;
  int wave = threadIdx.x >> 5;
  int m0 = blockIdx.y * 128 + (wave >> 1) * 32;
  int n0 = blockIdx.x * 64 + (wave & 1) * 32;
  v8f acc[2][2] = {};
  for (int k = 0; k < K; k += 32) {
    v16h a0 = load_a(A, K, m0,      M - 1, k, lane);
    v16h a1 = load_a(A, K, m0 + 16, M - 1, k, lane);
    v16h b0 = load_b(Bt, K, n0,      N - 1, k, lane);
    v16h b1 = load_b(Bt, K, n0 + 16, N - 1, k, lane);
    acc[0][0] = wmma16(a0, b0, acc[0][0]);
    acc[0][1] = wmma16(a0, b1, acc[0][1]);
    acc[1][0] = wmma16(a1, b0, acc[1][0]);
    acc[1][1] = wmma16(a1, b1, acc[1][1]);
  }
  int colb = lane & 15, rowb = (lane >> 4) * 8;
#pragma unroll
  for (int mi = 0; mi < 2; ++mi)
#pragma unroll
    for (int ni = 0; ni < 2; ++ni) {
      int gc = n0 + ni * 16 + colb;
      if (gc >= N) continue;
      float bv = bias ? bias[gc] : 0.f;
#pragma unroll
      for (int r = 0; r < 8; ++r) {
        int gm = m0 + mi * 16 + rowb + r;
        if (gm >= M) continue;
        float v = acc[mi][ni][r] + bv;
        if (act == 1) v = siluf(v);
        C[(size_t)gm * N + gc] = v;
      }
    }
}

// depthwise causal conv over xBC channels (W=4) + silu; emits fp32 AND f16 images
__global__ void k_conv_xbc(const float* __restrict__ zx, const float* __restrict__ w,
                           const float* __restrict__ bc, float* __restrict__ out,
                           _Float16* __restrict__ outh) {
  long idx = (long)blockIdx.x * blockDim.x + threadIdx.x; // B*SEQ*CONVD
  if (idx >= (long)B_ * SEQ_ * CONVD_) return;
  int c = (int)(idx % CONVD_);
  long rs = idx / CONVD_;
  int s = (int)(rs % SEQ_);
  long b = rs / SEQ_;
  float acc = bc[c];
#pragma unroll
  for (int kk = 0; kk < 4; ++kk) {
    int sp = s - 3 + kk;
    if (sp >= 0) acc += zx[(b * SEQ_ + sp) * DPROJ_ + DSSM_ + c] * w[c * 4 + kk];
  }
  float v = siluf(acc);
  out[idx] = v;
  outh[idx] = (_Float16)v;
}

__global__ void k_tmul(const float* __restrict__ a, const float* __restrict__ b,
                       float* __restrict__ o, int n) {
  int i = blockIdx.x * blockDim.x + threadIdx.x;
  if (i < n) o[i] = a[i] * b[i];
}

// depthwise causal conv over seq for (B*NH, SEQ) rows + silu
__global__ void k_tconv(const float* __restrict__ in, const float* __restrict__ w,
                        const float* __restrict__ bias, float* __restrict__ out) {
  long idx = (long)blockIdx.x * blockDim.x + threadIdx.x; // (B*NH)*SEQ
  if (idx >= (long)B_ * NH_ * SEQ_) return;
  int s = (int)(idx % SEQ_);
  int r = (int)(idx / SEQ_);
  int hh = r % NH_;
  float acc = bias[hh];
#pragma unroll
  for (int kk = 0; kk < 4; ++kk) {
    int sp = s - 3 + kk;
    if (sp >= 0) acc += in[(size_t)r * SEQ_ + sp] * w[hh * 4 + kk];
  }
  out[idx] = siluf(acc);
}

// dt_sp[b,h,s] = softplus(dt[b,s,h]*tdt[b,h,s] + dt_bias[h])
__global__ void k_dtsp(const float* __restrict__ zx, const float* __restrict__ tdt,
                       const float* __restrict__ dtb, float* __restrict__ dts) {
  long idx = (long)blockIdx.x * blockDim.x + threadIdx.x; // B*SEQ*NH, h fastest
  if (idx >= (long)B_ * SEQ_ * NH_) return;
  int h = (int)(idx % NH_);
  long rs = idx / NH_;           // b*SEQ + s
  int s = (int)(rs % SEQ_);
  long b = rs / SEQ_;
  float f = zx[rs * DPROJ_ + DSSM_ + CONVD_ + h] * tdt[(b * NH_ + h) * (size_t)SEQ_ + s] + dtb[h];
  float sp = (f > 20.f) ? f : log1pf(__expf(f));
  dts[(b * NH_ + h) * (size_t)SEQ_ + s] = sp;
}

// per (b,h,chunk) inclusive cumsum of a = -exp(A_log[h]) * dt
__global__ __launch_bounds__(256) void k_cumsum(const float* __restrict__ dts,
                                                const float* __restrict__ A_log,
                                                float* __restrict__ Acs) {
  int blk = blockIdx.x;          // bh*NC + c
  int c = blk % NC_;
  int bh = blk / NC_;
  int h = bh % NH_;
  int t = threadIdx.x;
  __shared__ float sc[CH_];
  float A = -__expf(A_log[h]);
  size_t base = (size_t)bh * SEQ_ + (size_t)c * CH_;
  sc[t] = A * dts[base + t];
  __syncthreads();
  for (int ofs = 1; ofs < CH_; ofs <<= 1) {
    float add = (t >= ofs) ? sc[t - ofs] : 0.f;
    __syncthreads();
    sc[t] += add;
    __syncthreads();
  }
  Acs[base + t] = sc[t];
}

// xT[b,h,p,s] and decay-prescaled xTw[b,h,p,s] = x * dt * exp(Acs_chunkend - Acs)
__global__ void k_xT(const float* __restrict__ xBCc, const float* __restrict__ dts,
                     const float* __restrict__ Acs, _Float16* __restrict__ xT,
                     _Float16* __restrict__ xTw) {
  long idx = (long)blockIdx.x * blockDim.x + threadIdx.x; // B*SEQ*DSSM
  if (idx >= (long)B_ * SEQ_ * DSSM_) return;
  int col = (int)(idx % DSSM_);
  long row = idx / DSSM_;        // b*SEQ + s
  int s = (int)(row % SEQ_);
  long b = row / SEQ_;
  int h = col / HD_, p = col % HD_;
  long bh = b * NH_ + h;
  float v = xBCc[row * CONVD_ + col];
  size_t dst = ((size_t)bh * HD_ + p) * SEQ_ + s;
  xT[dst] = (_Float16)v;
  float acs = Acs[(size_t)bh * SEQ_ + s];
  float ce  = Acs[(size_t)bh * SEQ_ + (s | (CH_ - 1))];
  xTw[dst] = (_Float16)(v * dts[(size_t)bh * SEQ_ + s] * __expf(ce - acs));
}

// BT[b,n,s] = Bm[b,s,n]
__global__ void k_BT(const float* __restrict__ xBCc, _Float16* __restrict__ BT) {
  long idx = (long)blockIdx.x * blockDim.x + threadIdx.x; // B*SEQ*128
  if (idx >= (long)B_ * SEQ_ * 128) return;
  int n = (int)(idx % 128);
  long row = idx / 128;          // b*SEQ + s
  int s = (int)(row % SEQ_);
  long b = row / SEQ_;
  BT[((size_t)b * 128 + n) * SEQ_ + s] = (_Float16)xBCc[row * CONVD_ + DSSM_ + n];
}

// Per (b,chunk,head): Ydiag = ((C*B^T) .* exp(segsum) .* dt) * x  (+ Dp*x), and
// chunk states S[p,n] = sum_z xw[p,z]*B[z,n]. The chunk's C/B tiles (64KB each)
// are staged into LDS once per workgroup by the Tensor Data Mover; all WMMA
// C/B fragment reads then hit LDS instead of VMEM.
__global__ __launch_bounds__(256) void k_ssd1(const _Float16* __restrict__ xh,
    const _Float16* __restrict__ xT, const _Float16* __restrict__ xTw,
    const _Float16* __restrict__ BT, const float* __restrict__ Acs,
    const float* __restrict__ dts, const float* __restrict__ xBCc,
    const float* __restrict__ Dp, float* __restrict__ Y, float* __restrict__ states) {
  int h = blockIdx.x, c = blockIdx.y, b = blockIdx.z;
  int bh = b * NH_ + h;
  int lane = threadIdx.x & 31, w = threadIdx.x >> 5, tid = threadIdx.x;
  __shared__ float sA[CH_], sd[CH_];
  __shared__ __align__(16) _Float16 mt[8][32 * 32];
  extern __shared__ __align__(16) _Float16 dynls[]; // 2 x 256x128 f16 = 128KB (<320KB WGP LDS)
  _Float16* sC = dynls;
  _Float16* sB = dynls + CH_ * 128;
  int sg = c * CH_;
  const _Float16* gC = xh + ((size_t)(b * SEQ_ + sg)) * CONVD_ + DSSM_ + 128;
  const _Float16* gB = xh + ((size_t)(b * SEQ_ + sg)) * CONVD_ + DSSM_;
  if (w == 0) {                                  // one wave drives the TDM
    tdm_load_2d((unsigned)(unsigned long long)sC, gC, CH_, 128, CONVD_);
    tdm_load_2d((unsigned)(unsigned long long)sB, gB, CH_, 128, CONVD_);
    __builtin_amdgcn_s_wait_tensorcnt(0);
  }
  sA[tid] = Acs[(size_t)bh * SEQ_ + sg + tid];
  sd[tid] = dts[(size_t)bh * SEQ_ + sg + tid];
  __syncthreads();                               // publish LDS tiles to all waves
  const _Float16* xTb = xT + (size_t)bh * HD_ * SEQ_;
  _Float16* mw = mt[w];
  int srow = 32 * w;
  int colb = lane & 15, rowb = (lane >> 4) * 8;
  v8f accy[2][4] = {};
  for (int zt = 0; zt <= w; ++zt) {           // only lower-triangular z-tiles
    int zrow = 32 * zt;
    v8f g[2][2] = {};
    for (int k = 0; k < 128; k += 32) {       // K = d_state, operands from LDS
      v16h a0 = load_a(sC, 128, srow,      255, k, lane);
      v16h a1 = load_a(sC, 128, srow + 16, 255, k, lane);
      v16h b0 = load_b(sB, 128, zrow,      255, k, lane);
      v16h b1 = load_b(sB, 128, zrow + 16, 255, k, lane);
      g[0][0] = wmma16(a0, b0, g[0][0]);
      g[0][1] = wmma16(a0, b1, g[0][1]);
      g[1][0] = wmma16(a1, b0, g[1][0]);
      g[1][1] = wmma16(a1, b1, g[1][1]);
    }
    // mask + decay + dt scaling, reshape C/D layout -> row-major LDS tile
#pragma unroll
    for (int mi = 0; mi < 2; ++mi)
#pragma unroll
      for (int ni = 0; ni < 2; ++ni) {
        int zl = zrow + ni * 16 + colb;
        float az = sA[zl], dz = sd[zl];
#pragma unroll
        for (int r = 0; r < 8; ++r) {
          int sl = srow + mi * 16 + rowb + r;
          float v = (zl <= sl) ? g[mi][ni][r] * __expf(sA[sl] - az) * dz : 0.f;
          mw[(sl - srow) * 32 + (zl - zrow)] = (_Float16)v;
        }
      }
    // per-wave tile; waves have divergent zt counts so no barrier -- LDS is
    // in-order within a wave, fence the compiler + drain DS counter.
    asm volatile("s_wait_dscnt 0" ::: "memory");
    v16h ma0 = load_a(mw, 32, 0,  31, 0, lane);
    v16h ma1 = load_a(mw, 32, 16, 31, 0, lane);
#pragma unroll
    for (int np = 0; np < 4; ++np) {
      v16h bb = load_b(xTb, SEQ_, np * 16, HD_ - 1, sg + zrow, lane);
      accy[0][np] = wmma16(ma0, bb, accy[0][np]);
      accy[1][np] = wmma16(ma1, bb, accy[1][np]);
    }
  }
  float dpv = Dp[h];
#pragma unroll
  for (int mi = 0; mi < 2; ++mi)
#pragma unroll
    for (int np = 0; np < 4; ++np)
#pragma unroll
      for (int r = 0; r < 8; ++r) {
        int sl = srow + mi * 16 + rowb + r;
        int p = np * 16 + colb;
        size_t row = (size_t)(b * SEQ_ + sg + sl);
        Y[row * DSSM_ + h * HD_ + p] = accy[mi][np][r] + dpv * xBCc[row * CONVD_ + h * HD_ + p];
      }
  // ---- chunk states: 64x128, K = 256, wave w -> p-tile (w>>1), n-tiles 4*(w&1).. ----
  const _Float16* xwb = xTw + (size_t)bh * HD_ * SEQ_;
  const _Float16* Btb = BT + (size_t)b * 128 * SEQ_;
  int mp = (w >> 1) * 16;
  int nb = (w & 1) * 4;
  v8f accs[4] = {};
  for (int k = 0; k < CH_; k += 32) {
    v16h aa = load_a(xwb, SEQ_, mp, HD_ - 1, sg + k, lane);
#pragma unroll
    for (int ni = 0; ni < 4; ++ni) {
      v16h bb = load_b(Btb, SEQ_, (nb + ni) * 16, 127, sg + k, lane);
      accs[ni] = wmma16(aa, bb, accs[ni]);
    }
  }
  float* st = states + ((size_t)bh * NC_ + c) * HD_ * 128;
#pragma unroll
  for (int ni = 0; ni < 4; ++ni)
#pragma unroll
    for (int r = 0; r < 8; ++r)
      st[(mp + rowb + r) * 128 + (nb + ni) * 16 + colb] = accs[ni][r];
}

// inter-chunk recurrence per (b,h): prev[c] = carry; carry = exp(Acs_end[c])*carry + states[c]
__global__ __launch_bounds__(256) void k_scan(const float* __restrict__ states,
                                              const float* __restrict__ Acs,
                                              _Float16* __restrict__ prevh) {
  int bh = blockIdx.x;
  int t = threadIdx.x;
  float carry[32];
#pragma unroll
  for (int j = 0; j < 32; ++j) carry[j] = 0.f;
  for (int c = 0; c < NC_; ++c) {
    size_t base = ((size_t)bh * NC_ + c) * (HD_ * 128);
#pragma unroll
    for (int j = 0; j < 32; ++j) prevh[base + t + 256 * j] = (_Float16)carry[j];
    float dc = __expf(Acs[(size_t)bh * SEQ_ + c * CH_ + (CH_ - 1)]);
#pragma unroll
    for (int j = 0; j < 32; ++j) carry[j] = dc * carry[j] + states[base + t + 256 * j];
  }
}

// Yoff: Y[s,p] += exp(Acs[s]) * sum_n C[s,n]*prev[p,n]
__global__ __launch_bounds__(256) void k_ssd3(const _Float16* __restrict__ xh,
    const _Float16* __restrict__ prevh, const float* __restrict__ Acs,
    float* __restrict__ Y) {
  int h = blockIdx.x, c = blockIdx.y, b = blockIdx.z;
  int bh = b * NH_ + h;
  int lane = threadIdx.x & 31, w = threadIdx.x >> 5, tid = threadIdx.x;
  __shared__ float sA[CH_];
  int sg = c * CH_;
  sA[tid] = Acs[(size_t)bh * SEQ_ + sg + tid];
  __syncthreads();
  const _Float16* Cc = xh + ((size_t)(b * SEQ_ + sg)) * CONVD_ + DSSM_ + 128;
  const _Float16* Pm = prevh + ((size_t)bh * NC_ + c) * (HD_ * 128);
  int srow = 32 * w;
  v8f acc[2][4] = {};
  for (int k = 0; k < 128; k += 32) {
    v16h a0 = load_a(Cc, CONVD_, srow,      255, k, lane);
    v16h a1 = load_a(Cc, CONVD_, srow + 16, 255, k, lane);
#pragma unroll
    for (int np = 0; np < 4; ++np) {
      v16h bb = load_b(Pm, 128, np * 16, HD_ - 1, k, lane);
      acc[0][np] = wmma16(a0, bb, acc[0][np]);
      acc[1][np] = wmma16(a1, bb, acc[1][np]);
    }
  }
  int colb = lane & 15, rowb = (lane >> 4) * 8;
#pragma unroll
  for (int mi = 0; mi < 2; ++mi)
#pragma unroll
    for (int np = 0; np < 4; ++np)
#pragma unroll
      for (int r = 0; r < 8; ++r) {
        int sl = srow + mi * 16 + rowb + r;
        size_t row = (size_t)(b * SEQ_ + sg + sl);
        Y[row * DSSM_ + h * HD_ + np * 16 + colb] += __expf(sA[sl]) * acc[mi][np][r];
      }
}

// gate with silu(z), RMS-norm over D_SSM, emit f16 activations for out_proj
__global__ __launch_bounds__(256) void k_gate_rms(const float* __restrict__ Y,
    const float* __restrict__ zx, const float* __restrict__ nw,
    _Float16* __restrict__ yh) {
  int row = blockIdx.x, t = threadIdx.x;
  __shared__ float red[256];
  const float* yr = Y + (size_t)row * DSSM_;
  const float* zr = zx + (size_t)row * DPROJ_;
  float vals[6];
  float ss = 0.f;
#pragma unroll
  for (int j = 0; j < 6; ++j) {
    int ci = t + 256 * j;
    float yg = yr[ci] * siluf(zr[ci]);
    vals[j] = yg; ss += yg * yg;
  }
  red[t] = ss; __syncthreads();
  for (int o = 128; o > 0; o >>= 1) { if (t < o) red[t] += red[t + o]; __syncthreads(); }
  float inv = rsqrtf(red[0] / (float)DSSM_ + 1e-12f);
#pragma unroll
  for (int j = 0; j < 6; ++j) {
    int ci = t + 256 * j;
    yh[(size_t)row * DSSM_ + ci] = (_Float16)(vals[j] * inv * nw[ci]);
  }
}

// time gate + mix + layernorm over SEQ, per (b,h) row
__global__ __launch_bounds__(256) void k_timeout(const float* __restrict__ td,
    const float* __restrict__ tdt, const float* __restrict__ gw,
    const float* __restrict__ gb, const float* __restrict__ lnw,
    const float* __restrict__ lnb, float* __restrict__ outp) {
  int bh = blockIdx.x, t = threadIdx.x;
  __shared__ float red[256];
  const float* tr = td + (size_t)bh * SEQ_;
  const float* dr = tdt + (size_t)bh * SEQ_;
  float dp = 0.f;
#pragma unroll
  for (int j = 0; j < 16; ++j) { int s = t + 256 * j; dp += tr[s] * gw[s]; }
  red[t] = dp; __syncthreads();
  for (int o = 128; o > 0; o >>= 1) { if (t < o) red[t] += red[t + o]; __syncthreads(); }
  float tg = 1.f / (1.f + __expf(-(red[0] + gb[0])));
  __syncthreads();
  float tv[16];
  float sum = 0.f, sq = 0.f;
#pragma unroll
  for (int j = 0; j < 16; ++j) {
    int s = t + 256 * j;
    float m = tg * tr[s] + (1.f - tg) * dr[s];
    tv[j] = m; sum += m; sq += m * m;
  }
  red[t] = sum; __syncthreads();
  for (int o = 128; o > 0; o >>= 1) { if (t < o) red[t] += red[t + o]; __syncthreads(); }
  float mu = red[0] / (float)SEQ_;
  __syncthreads();
  red[t] = sq; __syncthreads();
  for (int o = 128; o > 0; o >>= 1) { if (t < o) red[t] += red[t + o]; __syncthreads(); }
  float var = red[0] / (float)SEQ_ - mu * mu;
  float inv = rsqrtf(var + 1e-12f);
#pragma unroll
  for (int j = 0; j < 16; ++j) {
    int s = t + 256 * j;
    outp[(size_t)bh * SEQ_ + s] = (tv[j] - mu) * inv * lnw[s] + lnb[s];
  }
}

extern "C" void kernel_launch(void* const* d_in, const int* in_sizes, int n_in,
                              void* d_out, int out_size, void* d_ws, size_t ws_size,
                              hipStream_t stream) {
  (void)in_sizes; (void)n_in; (void)out_size; (void)ws_size;
  const float* u       = (const float*)d_in[0];
  const float* td      = (const float*)d_in[1];
  const float* W_in    = (const float*)d_in[2];
  const float* b_in    = (const float*)d_in[3];
  const float* conv_w  = (const float*)d_in[4];
  const float* conv_b  = (const float*)d_in[5];
  const float* tc1w    = (const float*)d_in[6];
  const float* tc1b    = (const float*)d_in[7];
  const float* tc2w    = (const float*)d_in[8];
  const float* tc2b    = (const float*)d_in[9];
  const float* w1      = (const float*)d_in[10];
  const float* b1      = (const float*)d_in[11];
  const float* w2      = (const float*)d_in[12];
  const float* b2      = (const float*)d_in[13];
  const float* gate_w  = (const float*)d_in[14];
  const float* gate_b  = (const float*)d_in[15];
  const float* ln_w    = (const float*)d_in[16];
  const float* ln_b    = (const float*)d_in[17];
  const float* A_log   = (const float*)d_in[18];
  const float* dt_bias = (const float*)d_in[19];
  const float* Dp      = (const float*)d_in[20];
  const float* norm_w  = (const float*)d_in[21];
  const float* W_out   = (const float*)d_in[22];
  const float* b_out   = (const float*)d_in[23];

  char* base = (char*)d_ws;
  size_t off = 0;
  auto alloc_h = [&](size_t n) { _Float16* p = (_Float16*)(base + off);
                                 off = (off + n * 2 + 255) & ~(size_t)255; return p; };
  auto alloc_f = [&](size_t n) { float* p = (float*)(base + off);
                                 off = (off + n * 4 + 255) & ~(size_t)255; return p; };

  const size_t MS = (size_t)B_ * SEQ_;           // 8192 token rows
  _Float16* u_h    = alloc_h(MS * DMODEL_);
  _Float16* Win_h  = alloc_h((size_t)DPROJ_ * DMODEL_);
  _Float16* w1_h   = alloc_h((size_t)SEQ_ * SEQ_);
  _Float16* w2_h   = alloc_h((size_t)SEQ_ * SEQ_);
  _Float16* Wout_h = alloc_h((size_t)DMODEL_ * DSSM_);
  _Float16* td_h   = alloc_h((size_t)B_ * NH_ * SEQ_);
  float*    zx     = alloc_f(MS * DPROJ_);
  float*    xBCc   = alloc_f(MS * CONVD_);
  _Float16* xh     = alloc_h(MS * CONVD_);
  float*    ts1    = alloc_f((size_t)B_ * NH_ * SEQ_);
  _Float16* ts1_h  = alloc_h((size_t)B_ * NH_ * SEQ_);
  float*    ts     = alloc_f((size_t)B_ * NH_ * SEQ_);
  float*    tdt0   = alloc_f((size_t)B_ * NH_ * SEQ_);
  float*    tdt1   = alloc_f((size_t)B_ * NH_ * SEQ_);
  float*    dts    = alloc_f((size_t)B_ * NH_ * SEQ_);
  float*    Acs    = alloc_f((size_t)B_ * NH_ * SEQ_);
  _Float16* xT_h   = alloc_h(MS * DSSM_);
  _Float16* xTw_h  = alloc_h(MS * DSSM_);
  _Float16* BT_h   = alloc_h((size_t)B_ * 128 * SEQ_);
  float*    states = alloc_f((size_t)B_ * NH_ * NC_ * HD_ * 128);
  _Float16* prev_h = alloc_h((size_t)B_ * NH_ * NC_ * HD_ * 128);
  float*    Ybuf   = alloc_f(MS * DSSM_);
  _Float16* y_h    = alloc_h(MS * DSSM_);

  float* out_main = (float*)d_out;                       // (B,SEQ,D_MODEL)
  float* out_time = (float*)d_out + MS * DMODEL_;        // (B,NH,SEQ)

  // fp32 -> f16 operand staging
  k_cvt<<<2048, 256, 0, stream>>>(u, u_h, (long)MS * DMODEL_);
  k_cvt<<<2048, 256, 0, stream>>>(W_in, Win_h, (long)DPROJ_ * DMODEL_);
  k_cvt<<<4096, 256, 0, stream>>>(w1, w1_h, (long)SEQ_ * SEQ_);
  k_cvt<<<4096, 256, 0, stream>>>(w2, w2_h, (long)SEQ_ * SEQ_);
  k_cvt<<<1024, 256, 0, stream>>>(W_out, Wout_h, (long)DMODEL_ * DSSM_);
  k_cvt<<<256, 256, 0, stream>>>(td, td_h, (long)B_ * NH_ * SEQ_);

  // in_proj: (8192 x 3352) = u * W_in^T + b_in
  k_gemm<<<dim3((DPROJ_ + 63) / 64, (int)(MS + 127) / 128), 256, 0, stream>>>(
      u_h, Win_h, b_in, zx, (int)MS, DPROJ_, DMODEL_, 0);

  // conv + silu on xBC slice; fused fp32 + f16 outputs
  k_conv_xbc<<<(int)(MS * CONVD_ / 256), 256, 0, stream>>>(zx, conv_w, conv_b, xBCc, xh);

  // time MLP: 48 x 4096 rows, K=4096, silu on first
  k_gemm<<<dim3(64, 1), 256, 0, stream>>>(td_h, w1_h, b1, ts1, B_ * NH_, SEQ_, SEQ_, 1);
  k_cvt<<<256, 256, 0, stream>>>(ts1, ts1_h, (long)B_ * NH_ * SEQ_);
  k_gemm<<<dim3(64, 1), 256, 0, stream>>>(ts1_h, w2_h, b2, ts, B_ * NH_, SEQ_, SEQ_, 0);
  k_tmul<<<768, 256, 0, stream>>>(ts, td, tdt0, B_ * NH_ * SEQ_);
  k_tconv<<<768, 256, 0, stream>>>(tdt0, tc1w, tc1b, tdt1);
  k_tconv<<<768, 256, 0, stream>>>(tdt1, tc2w, tc2b, tdt0);   // final tdt in tdt0

  // dt softplus, per-chunk cumsum, transposed/prescaled operand staging
  k_dtsp<<<768, 256, 0, stream>>>(zx, tdt0, dt_bias, dts);
  k_cumsum<<<B_ * NH_ * NC_, 256, 0, stream>>>(dts, A_log, Acs);
  k_xT<<<(int)(MS * DSSM_ / 256), 256, 0, stream>>>(xBCc, dts, Acs, xT_h, xTw_h);
  k_BT<<<(int)(MS * 128 / 256), 256, 0, stream>>>(xBCc, BT_h);

  // SSD: diag block + chunk states (TDM-staged LDS tiles) -> scan -> off-diag
  k_ssd1<<<dim3(NH_, NC_, B_), 256, 2 * CH_ * 128 * sizeof(_Float16), stream>>>(
      xh, xT_h, xTw_h, BT_h, Acs, dts, xBCc, Dp, Ybuf, states);
  k_scan<<<B_ * NH_, 256, 0, stream>>>(states, Acs, prev_h);
  k_ssd3<<<dim3(NH_, NC_, B_), 256, 0, stream>>>(xh, prev_h, Acs, Ybuf);

  // gate + RMS norm, out_proj, time-path output
  k_gate_rms<<<(int)MS, 256, 0, stream>>>(Ybuf, zx, norm_w, y_h);
  k_gemm<<<dim3(DMODEL_ / 64, (int)(MS / 128)), 256, 0, stream>>>(
      y_h, Wout_h, b_out, out_main, (int)MS, DMODEL_, DSSM_, 0);
  k_timeout<<<B_ * NH_, 256, 0, stream>>>(td, tdt0, gate_w, gate_b, ln_w, ln_b, out_time);
}
